// vhgae_encoder_6803228196945
// MI455X (gfx1250) — compile-verified, matrix operand order checked by reading, not run on hardware
//
#include <hip/hip_runtime.h>
#include <hip/hip_bf16.h>
#include <math.h>

typedef __attribute__((ext_vector_type(16))) __bf16 v16bf;
typedef __attribute__((ext_vector_type(8)))  float  v8f;

#define DMODEL 512
#define TILE_M 128
#define TILE_N 128
#define TILE_K 32
#define NKT (DMODEL / TILE_K)        // 16 K-steps
#define LDS_STRIDE (TILE_K + 8)      // 40 bf16 = 80B rows -> 16B-aligned 8-elem chunks

// ---------------------------------------------------------------------------
// utility kernels
// ---------------------------------------------------------------------------
__global__ void zero_f32_kernel(float* __restrict__ p, long n) {
  long i = (long)blockIdx.x * blockDim.x + threadIdx.x;
  if (i < n) p[i] = 0.0f;
}

__global__ void f32_to_bf16_kernel(const float* __restrict__ in,
                                   __bf16* __restrict__ out, long n) {
  long i = (long)blockIdx.x * blockDim.x + threadIdx.x;
  if (i < n) out[i] = (__bf16)in[i];
}

__global__ void init_min_kernel(int* p) { *p = 0x7fffffff; }

__global__ void min_reduce_kernel(const int* __restrict__ idx, int n, int* p) {
  int i = blockIdx.x * blockDim.x + threadIdx.x;
  if (i < n) atomicMin(p, idx[i]);
}

__global__ void seg_count_kernel(const int* __restrict__ seg,
                                 const int* __restrict__ dmin, int useShift,
                                 float* __restrict__ cnt, int E) {
  int e = blockIdx.x * blockDim.x + threadIdx.x;
  if (e < E) {
    int s = seg[e] - (useShift ? *dmin : 0);
    atomicAdd(&cnt[s], 1.0f);
  }
}

// agg[sidx[e]] += norm[e] * t[gidx[e]]   (per 4-float chunk per thread)
__global__ void scatter_kernel(const float* __restrict__ t,
                               const int* __restrict__ gidx, int shiftG,
                               const int* __restrict__ sidx, int shiftS,
                               const int* __restrict__ dmin,
                               const float* __restrict__ norm,
                               float* __restrict__ agg, int E) {
  long gid = (long)blockIdx.x * blockDim.x + threadIdx.x;
  if (gid >= (long)E * (DMODEL / 4)) return;
  int e = (int)(gid >> 7);
  int c = ((int)gid & 127) << 2;
  int dm = *dmin;
  int g = gidx[e] - (shiftG ? dm : 0);
  int s = sidx[e] - (shiftS ? dm : 0);
  float nv = norm[e];
  const float4 v = *(const float4*)(t + (size_t)g * DMODEL + c);
  float* d = agg + (size_t)s * DMODEL + c;
  atomicAdd(d + 0, nv * v.x);
  atomicAdd(d + 1, nv * v.y);
  atomicAdd(d + 2, nv * v.z);
  atomicAdd(d + 3, nv * v.w);
}

// std = softplus(stdraw) in place; final = noise*std + mean
__global__ void vae_combine_kernel(float* __restrict__ stdbuf,
                                   const float* __restrict__ meanbuf,
                                   const float* __restrict__ noise,
                                   float* __restrict__ finalbuf, long n) {
  long i = (long)blockIdx.x * blockDim.x + threadIdx.x;
  if (i < n) {
    float xv = stdbuf[i];
    float sp = fmaxf(xv, 0.0f) + log1pf(expf(-fabsf(xv)));
    stdbuf[i] = sp;
    finalbuf[i] = noise[i] * sp + meanbuf[i];
  }
}

// ---------------------------------------------------------------------------
// LayerNorm (fused optional segment-mean divide) -> bf16 output
// one block (256 threads) per row of 512; single fused (sum, sumsq) reduction
// ---------------------------------------------------------------------------
__global__ __launch_bounds__(256)
void ln_to_bf16_kernel(const float* __restrict__ in, const float* __restrict__ cnt,
                       const float* __restrict__ w, const float* __restrict__ b,
                       __bf16* __restrict__ out, int rows) {
  __shared__ float r1[256];
  __shared__ float r2[256];
  const int row = blockIdx.x;
  const int tid = threadIdx.x;
  if (row >= rows) return;
  float scale = 1.0f;
  if (cnt) scale = 1.0f / fmaxf(cnt[row], 1.0f);
  const size_t base = (size_t)row * DMODEL;
  float v0 = in[base + tid] * scale;
  float v1 = in[base + tid + 256] * scale;
  r1[tid] = v0 + v1;
  r2[tid] = v0 * v0 + v1 * v1;
  __syncthreads();
  for (int st = 128; st > 0; st >>= 1) {
    if (tid < st) { r1[tid] += r1[tid + st]; r2[tid] += r2[tid + st]; }
    __syncthreads();
  }
  float mu  = r1[0] * (1.0f / DMODEL);
  float var = r2[0] * (1.0f / DMODEL) - mu * mu;
  float rstd = rsqrtf(var + 1e-5f);
  out[base + tid]       = (__bf16)((v0 - mu) * rstd * w[tid] + b[tid]);
  out[base + tid + 256] = (__bf16)((v1 - mu) * rstd * w[tid + 256] + b[tid + 256]);
}

// ---------------------------------------------------------------------------
// GEMM: C[M,512] = act(A[M,512](bf16) @ W[512,512](bf16) + bias), f32 out.
// 256 threads = 8 wave32 waves arranged 4(M) x 2(N); block tile 128x128,
// K-step 32, double-buffered LDS. Each wave owns a 32x64 strip -> 8x
// v_wmma_f32_16x16x32_bf16 per K-step with B-fragment reuse. A tile is copied
// global->LDS with gfx1250 global_load_async_to_lds_b128 (ASYNCcnt).
// ---------------------------------------------------------------------------
__global__ __launch_bounds__(256)
void gemm_bf16_kernel(const __bf16* __restrict__ A, const __bf16* __restrict__ W,
                      const float* __restrict__ bias, float* __restrict__ C,
                      int M, int relu) {
  __shared__ __bf16 As[2][TILE_M][LDS_STRIDE];
  __shared__ __bf16 Bs[2][TILE_N][LDS_STRIDE];   // transposed: Bs[b][n][k]

  const int tid  = threadIdx.x;
  const int lane = tid & 31;
  const int wave = tid >> 5;          // 8 wave32 waves
  const int wm   = wave & 3;          // 4 M-strips of 32 rows
  const int wn   = wave >> 2;         // 2 N-strips of 64 cols
  const int half = (lane >> 4) & 1;
  const int r    = lane & 15;
  const int n0   = blockIdx.x * TILE_N;
  const int m0   = blockIdx.y * TILE_M;
  const int Mm1  = M - 1;

  v8f acc[2][4];
  {
    v8f vzero = {0.f, 0.f, 0.f, 0.f, 0.f, 0.f, 0.f, 0.f};
#pragma unroll
    for (int mt = 0; mt < 2; ++mt)
#pragma unroll
      for (int nt = 0; nt < 4; ++nt) acc[mt][nt] = vzero;
  }

  // A tile: 128x32 bf16 -> 512x 16B slots; 2 async b128 copies per thread.
  auto loadA = [&](int kk, int buf) {
#pragma unroll
    for (int it = 0; it < 2; ++it) {
      int slot = tid + it * 256;
      int row  = slot >> 2;
      int kc   = (slot & 3) * 8;
      int gr   = m0 + row;
      if (gr > Mm1) gr = Mm1;                       // clamp: no EXEC divergence
      unsigned long long ga = (unsigned long long)(A + (size_t)gr * DMODEL + kk + kc);
      unsigned lo = (unsigned)(uintptr_t)(&As[buf][row][kc]);
      asm volatile("global_load_async_to_lds_b128 %0, %1, off"
                   :: "v"(lo), "v"(ga) : "memory");
    }
  };
  // B tile: W[kk..+32, n0..+128] -> transposed Bs[n][k]
  auto loadB = [&](int kk, int buf) {
#pragma unroll
    for (int it = 0; it < 2; ++it) {
      int slot = tid + it * 256;      // 512 slots = 32 k x 16 n-chunks
      int k  = slot >> 4;
      int nc = (slot & 15) * 8;
      uint4 v = *(const uint4*)(W + (size_t)(kk + k) * DMODEL + n0 + nc);
      const __bf16* pv = (const __bf16*)&v;
#pragma unroll
      for (int j = 0; j < 8; ++j) Bs[buf][nc + j][k] = pv[j];
    }
  };

  loadA(0, 0);
  loadB(0, 0);
  asm volatile("s_wait_asynccnt 0x0" ::: "memory");
  __syncthreads();

  for (int kt = 0; kt < NKT; ++kt) {
    const int cur = kt & 1;
    if (kt + 1 < NKT) {
      loadA((kt + 1) * TILE_K, cur ^ 1);
      loadB((kt + 1) * TILE_K, cur ^ 1);
      __builtin_prefetch(W + (size_t)((kt + 1) * TILE_K + (tid >> 4)) * DMODEL + n0, 0, 0);
    }

    // fragments from the current buffer (ISA 16-bit layouts)
    v16bf af[2];
#pragma unroll
    for (int mt = 0; mt < 2; ++mt) {
      const __bf16* arow = &As[cur][wm * 32 + mt * 16 + r][0];
#pragma unroll
      for (int v = 0; v < 8; ++v) {
        int kp = ((v & 3) * 2) + ((v >> 2) * 16) + half * 8;
        af[mt][2 * v]     = arow[kp];
        af[mt][2 * v + 1] = arow[kp + 1];
      }
    }
    v16bf bf4[4];
#pragma unroll
    for (int nt = 0; nt < 4; ++nt) {
      const __bf16* bcol = &Bs[cur][wn * 64 + nt * 16 + r][0];
#pragma unroll
      for (int v = 0; v < 8; ++v) {
        int kp = ((v & 3) * 2) + ((v >> 2) * 16) + half * 8;
        bf4[nt][2 * v]     = bcol[kp];
        bf4[nt][2 * v + 1] = bcol[kp + 1];
      }
    }
#pragma unroll
    for (int mt = 0; mt < 2; ++mt)
#pragma unroll
      for (int nt = 0; nt < 4; ++nt)
        acc[mt][nt] = __builtin_amdgcn_wmma_f32_16x16x32_bf16(
            false, af[mt], false, bf4[nt], (short)0, acc[mt][nt], false, false);

    asm volatile("s_wait_asynccnt 0x0" ::: "memory");
    __syncthreads();
  }

  // epilogue: C/D layout -> m = vgpr + 8*half, n = lane&15
#pragma unroll
  for (int mt = 0; mt < 2; ++mt) {
#pragma unroll
    for (int nt = 0; nt < 4; ++nt) {
      int col = n0 + wn * 64 + nt * 16 + r;
      float bv = bias[col];
#pragma unroll
      for (int v = 0; v < 8; ++v) {
        int grow = m0 + wm * 32 + mt * 16 + v + half * 8;
        if (grow < M) {
          float val = acc[mt][nt][v] + bv;
          if (relu) val = fmaxf(val, 0.0f);
          C[(size_t)grow * DMODEL + col] = val;
        }
      }
    }
  }
}

// ---------------------------------------------------------------------------
// host-side helpers (all launches on `stream`; graph-capture safe)
// ---------------------------------------------------------------------------
static void run_mlp(const float* in, const float* cnt, int rows,
                    const float* ln0w, const float* ln0b,
                    const __bf16* W1, const float* b1,
                    const float* ln1w, const float* ln1b,
                    const __bf16* W2, const float* b2,
                    float* out, __bf16* Abf, hipStream_t stream) {
  dim3 gg(DMODEL / TILE_N, (rows + TILE_M - 1) / TILE_M);
  ln_to_bf16_kernel<<<rows, 256, 0, stream>>>(in, cnt, ln0w, ln0b, Abf, rows);
  gemm_bf16_kernel<<<gg, 256, 0, stream>>>(Abf, W1, b1, out, rows, 1);
  ln_to_bf16_kernel<<<rows, 256, 0, stream>>>(out, nullptr, ln1w, ln1b, Abf, rows);
  gemm_bf16_kernel<<<gg, 256, 0, stream>>>(Abf, W2, b2, out, rows, 1);  // outer relu
}

static void run_head(const float* z, int rows,
                     const __bf16* W1, const float* b1,
                     const __bf16* W2, const float* b2,
                     float* tmp, __bf16* Abf, float* out, hipStream_t stream) {
  long n = (long)rows * DMODEL;
  int cb = (int)((n + 255) / 256);
  dim3 gg(DMODEL / TILE_N, (rows + TILE_M - 1) / TILE_M);
  f32_to_bf16_kernel<<<cb, 256, 0, stream>>>(z, Abf, n);
  gemm_bf16_kernel<<<gg, 256, 0, stream>>>(Abf, W1, b1, tmp, rows, 1);
  f32_to_bf16_kernel<<<cb, 256, 0, stream>>>(tmp, Abf, n);
  gemm_bf16_kernel<<<gg, 256, 0, stream>>>(Abf, W2, b2, out, rows, 0);
}

extern "C" void kernel_launch(void* const* d_in, const int* in_sizes, int n_in,
                              void* d_out, int out_size, void* d_ws, size_t ws_size,
                              hipStream_t stream) {
  (void)n_in; (void)out_size; (void)ws_size;
  const float* x        = (const float*)d_in[0];
  const int*   ei       = (const int*)d_in[1];
  const float* norm     = (const float*)d_in[2];
  const float* ln0w_all = (const float*)d_in[3];
  const float* ln0b_all = (const float*)d_in[4];
  const float* W1_all   = (const float*)d_in[5];
  const float* b1_all   = (const float*)d_in[6];
  const float* ln1w_all = (const float*)d_in[7];
  const float* ln1b_all = (const float*)d_in[8];
  const float* W2_all   = (const float*)d_in[9];
  const float* b2_all   = (const float*)d_in[10];
  const float* encW1    = (const float*)d_in[11];
  const float* encb1    = (const float*)d_in[12];
  const float* encW2    = (const float*)d_in[13];
  const float* encb2    = (const float*)d_in[14];
  const float* noise_n  = (const float*)d_in[15];
  const float* noise_h  = (const float*)d_in[16];
  float* out = (float*)d_out;

  const int Nn = in_sizes[0] / DMODEL;             // 50000
  const int Nh = in_sizes[16] / DMODEL;            // 10000
  const int E  = in_sizes[2];                      // 250000
  const int* src    = ei;                          // edge_index[0]
  const int* dstArr = ei + E;                      // edge_index[1]
  const size_t MS = (size_t)DMODEL * DMODEL;       // 262144

  // ---- workspace carve ----
  size_t off = 0;
  auto carve = [&](size_t bytes) -> void* {
    void* p = (char*)d_ws + off;
    off = (off + bytes + 255) & ~(size_t)255;
    return p;
  };
  __bf16* wMLP1 = (__bf16*)carve(8 * MS * sizeof(__bf16));
  __bf16* wMLP2 = (__bf16*)carve(8 * MS * sizeof(__bf16));
  __bf16* wENC1 = (__bf16*)carve(4 * MS * sizeof(__bf16));
  __bf16* wENC2 = (__bf16*)carve(4 * MS * sizeof(__bf16));
  __bf16* Abf   = (__bf16*)carve((size_t)Nn * DMODEL * sizeof(__bf16));
  float* NB0 = (float*)carve((size_t)Nn * DMODEL * sizeof(float));
  float* NB1 = (float*)carve((size_t)Nn * DMODEL * sizeof(float));
  float* NB2 = (float*)carve((size_t)Nn * DMODEL * sizeof(float));
  float* HB0 = (float*)carve((size_t)Nh * DMODEL * sizeof(float));
  float* HB1 = (float*)carve((size_t)Nh * DMODEL * sizeof(float));
  float* HB2 = (float*)carve((size_t)Nh * DMODEL * sizeof(float));
  float* cntN = (float*)carve((size_t)Nn * sizeof(float));
  float* cntH = (float*)carve((size_t)Nh * sizeof(float));
  int*   dmin = (int*)carve(sizeof(int));

  // ---- weights -> bf16 ----
  {
    long n1 = (long)8 * MS, n2 = (long)4 * MS;
    f32_to_bf16_kernel<<<(int)((n1 + 255) / 256), 256, 0, stream>>>(W1_all, wMLP1, n1);
    f32_to_bf16_kernel<<<(int)((n1 + 255) / 256), 256, 0, stream>>>(W2_all, wMLP2, n1);
    f32_to_bf16_kernel<<<(int)((n2 + 255) / 256), 256, 0, stream>>>(encW1, wENC1, n2);
    f32_to_bf16_kernel<<<(int)((n2 + 255) / 256), 256, 0, stream>>>(encW2, wENC2, n2);
  }
  // ---- dst index shift (min of edge_index[1]) ----
  init_min_kernel<<<1, 1, 0, stream>>>(dmin);
  min_reduce_kernel<<<(E + 255) / 256, 256, 0, stream>>>(dstArr, E, dmin);

  const int scat_blocks = (int)(((long)E * (DMODEL / 4) + 255) / 256);
  auto mlp = [&](int k, const float* in, const float* cnt, int rows, float* o) {
    run_mlp(in, cnt, rows,
            ln0w_all + (size_t)k * DMODEL, ln0b_all + (size_t)k * DMODEL,
            wMLP1 + (size_t)k * MS, b1_all + (size_t)k * DMODEL,
            ln1w_all + (size_t)k * DMODEL, ln1b_all + (size_t)k * DMODEL,
            wMLP2 + (size_t)k * MS, b2_all + (size_t)k * DMODEL,
            o, Abf, stream);
  };

  for (int i = 0; i < 2; ++i) {
    const float* hin = (i == 0) ? x : NB2;
    const int mB = 4 * i;
    // ---- V2E: enc on nodes, scatter to hyperedges, dec on hyperedges ----
    mlp(mB + 0, hin, nullptr, Nn, NB0);                       // tN = relu(mlp_enc(h))
    zero_f32_kernel<<<(int)(((long)Nh * DMODEL + 255) / 256), 256, 0, stream>>>(HB0, (long)Nh * DMODEL);
    zero_f32_kernel<<<(Nh + 255) / 256, 256, 0, stream>>>(cntH, Nh);
    seg_count_kernel<<<(E + 255) / 256, 256, 0, stream>>>(dstArr, dmin, 1, cntH, E);
    scatter_kernel<<<scat_blocks, 256, 0, stream>>>(NB0, src, 0, dstArr, 1, dmin, norm, HB0, E);
    mlp(mB + 1, HB0, cntH, Nh, HB1);                          // x_he (mean divide fused in LN)
    // ---- E2V: enc on hyperedges, scatter to nodes, dec on nodes ----
    mlp(mB + 2, HB1, nullptr, Nh, HB2);                       // tH
    zero_f32_kernel<<<(int)(((long)Nn * DMODEL + 255) / 256), 256, 0, stream>>>(NB1, (long)Nn * DMODEL);
    zero_f32_kernel<<<(Nn + 255) / 256, 256, 0, stream>>>(cntN, Nn);
    seg_count_kernel<<<(E + 255) / 256, 256, 0, stream>>>(src, dmin, 0, cntN, E);
    scatter_kernel<<<scat_blocks, 256, 0, stream>>>(HB2, dstArr, 1, src, 0, dmin, norm, NB1, E);
    mlp(mB + 3, NB1, cntN, Nn, NB2);                          // x_node
  }

  // ---- heads: x_node = NB2, x_he = HB1 ----
  const size_t SN = (size_t)Nn * DMODEL, SH = (size_t)Nh * DMODEL;
  float* out_final_n = out;
  float* out_mean_n  = out + SN;
  float* out_std_n   = out + 2 * SN;
  float* out_final_h = out + 3 * SN;
  float* out_mean_h  = out + 3 * SN + SH;
  float* out_std_h   = out + 3 * SN + 2 * SH;

  run_head(NB2, Nn, wENC1 + 0 * MS, encb1 + 0 * DMODEL, wENC2 + 0 * MS, encb2 + 0 * DMODEL,
           NB0, Abf, out_mean_n, stream);
  run_head(NB2, Nn, wENC1 + 1 * MS, encb1 + 1 * DMODEL, wENC2 + 1 * MS, encb2 + 1 * DMODEL,
           NB0, Abf, out_std_n, stream);
  vae_combine_kernel<<<(int)((SN + 255) / 256), 256, 0, stream>>>(out_std_n, out_mean_n,
                                                                  noise_n, out_final_n, (long)SN);
  run_head(HB1, Nh, wENC1 + 2 * MS, encb1 + 2 * DMODEL, wENC2 + 2 * MS, encb2 + 2 * DMODEL,
           HB0, Abf, out_mean_h, stream);
  run_head(HB1, Nh, wENC1 + 3 * MS, encb1 + 3 * DMODEL, wENC2 + 3 * MS, encb2 + 3 * DMODEL,
           HB0, Abf, out_std_h, stream);
  vae_combine_kernel<<<(int)((SH + 255) / 256), 256, 0, stream>>>(out_std_h, out_mean_h,
                                                                  noise_h, out_final_h, (long)SH);
}
